// Vocoder_68891275428317
// MI455X (gfx1250) — compile-verified, matrix operand order checked by reading, not run on hardware
//
#include <hip/hip_runtime.h>
#include <hip/hip_bf16.h>

// ---------------------------------------------------------------------------
// WaveRNN-style vocoder for MI455X (gfx1250, wave32, WMMA).
//
// The GRU scan is over N=8 steps with batch T=8192, so every step is a big
// GEMM (8192x1536x512). All GEMMs go through v_wmma_f32_16x16x32_f16
// (f16 inputs, f32 accumulate). Tiles are staged into LDS with the CDNA5
// async DMA path (global_load_async_to_lds_b128, ASYNCcnt) so staging needs
// no VGPR round-trip; per-wave 32x64 accumulator tile (8 WMMA / K-chunk,
// A reused 4x, B reused 2x). ~490 GFLOP vs ~1.5 GB HBM -> compute-leaning.
// ---------------------------------------------------------------------------

#define N_STEPS 8
#define T_BATCH 8192
#define HID 512
#define FEAT 128
#define KIN 160          // 129 padded up to a multiple of 32
#define G3 (3 * HID)     // 1536
#define NCLASS 512
#define NT (N_STEPS * T_BATCH)  // 65536

typedef __attribute__((ext_vector_type(8)))  _Float16 v8h;
typedef __attribute__((ext_vector_type(16))) _Float16 v16h;
typedef __attribute__((ext_vector_type(8)))  float    v8f;

union Frag16 { v16h v; v8h h[2]; };

// ---------------------------------------------------------------------------
// WMMA helper: D = A(16x32 f16) * B(32x16 f16) + C(16x16 f32)
// ---------------------------------------------------------------------------
__device__ __forceinline__ v8f wmma_f32_f16(v16h a, v16h b, v8f c) {
  return __builtin_amdgcn_wmma_f32_16x16x32_f16(
      /*neg_a=*/false, a, /*neg_b=*/false, b,
      /*c_mod=*/(short)0, c, /*reuse_a=*/false, /*reuse_b=*/false);
}

// Async DMA: 16 bytes per lane, global -> LDS, tracked by ASYNCcnt.
// INST_OFFSET is added to BOTH the LDS and the global address (ISA 08 §4.4),
// so one base pair + offset immediates stages a contiguous row on both sides.
#define ASYNC_LDS_B128(ldsOff, gptr, IMMOFF)                                   \
  asm volatile("global_load_async_to_lds_b128 %0, %1, off offset:" #IMMOFF     \
               :: "v"(ldsOff), "v"(gptr) : "memory")

__device__ __forceinline__ void wait_asynccnt0() {
  asm volatile("s_wait_asynccnt 0x0" ::: "memory");
}

// Load a 16x32 f16 fragment from LDS (row-major, row stride `ldr` halves).
// ISA layout: lane L<16 -> row M=L, K = {0..7, 16..23}; lane L>=16 -> row
// M=L-16, K = {8..15, 24..31}. All accesses 16B aligned (ldr*2 % 16 == 0).
__device__ __forceinline__ v16h lds_frag(const _Float16* base, int ldr) {
  const int lane = threadIdx.x & 31;
  const int r    = lane & 15;
  const int ks   = (lane >> 4) << 3;   // 0 or 8
  const _Float16* p = base + r * ldr + ks;
  Frag16 f;
  f.h[0] = *(const v8h*)(p);        // K = ks .. ks+7
  f.h[1] = *(const v8h*)(p + 16);   // K = ks+16 .. ks+23
  return f.v;
}

// ---------------------------------------------------------------------------
// Tiled WMMA GEMM:  C(M,Nc) = act( A_f16(M,K) @ W_f16(Nc,K)^T + bias )
//   block = 256 threads = 8 waves (2 in M x 4 in N); block tile 64 x 256.
//   Each wave: 2x4 grid of 16x16 f32 accumulators (8 WMMA per K chunk).
//   Requires: M % 64 == 0, Nc % 256 == 0, K % 32 == 0 (true for all uses).
// ---------------------------------------------------------------------------
#define BM 64
#define BN 256
#define KC 32
#define LDSS 40   // padded LDS row stride (halves); 80B, 16B-aligned rows

__global__ __launch_bounds__(256)
void k_gemm_f16(const _Float16* __restrict__ A,
                const _Float16* __restrict__ W,
                const float* __restrict__ bias,
                float* __restrict__ C32,          // nullable
                _Float16* __restrict__ C16,       // nullable
                int M, int Nc, int K, int relu) {
  __shared__ _Float16 sA[BM * LDSS];   //  5.0 KB
  __shared__ _Float16 sB[BN * LDSS];   // 20.0 KB

  const int tid  = threadIdx.x;
  const int lane = tid & 31;
  const int wave = tid >> 5;
  const int wm   = wave & 1;    // wave's 32-row slab in M
  const int wn   = wave >> 1;   // wave's 64-col slab in N
  const int bm   = blockIdx.y * BM;
  const int bn   = blockIdx.x * BN;

  // Staging addressing: A: 64 rows x 32 halves (one b128/thread);
  //                     B: one full 32-half row per thread (4 b128).
  const int arow = tid >> 2, acg = (tid & 3) * 8;
  const _Float16* gA = A + (size_t)(bm + arow) * K + acg;
  const _Float16* gB = W + (size_t)(bn + tid) * K;
  const unsigned ldsA = (unsigned)(uintptr_t)&sA[arow * LDSS + acg];
  const unsigned ldsB = (unsigned)(uintptr_t)&sB[tid * LDSS];

  v8f acc[2][4];
#pragma unroll
  for (int i = 0; i < 2; ++i)
#pragma unroll
    for (int j = 0; j < 4; ++j) acc[i][j] = (v8f){};

  for (int kt = 0; kt < K; kt += KC) {
    // Async stage this K-chunk into LDS (no VGPR round-trip).
    ASYNC_LDS_B128(ldsA, gA + kt, 0);
    ASYNC_LDS_B128(ldsB, gB + kt, 0);
    ASYNC_LDS_B128(ldsB, gB + kt, 16);
    ASYNC_LDS_B128(ldsB, gB + kt, 32);
    ASYNC_LDS_B128(ldsB, gB + kt, 48);
    if (kt + KC < K) {  // warm L2 for the next chunk -> global_prefetch_b8
      __builtin_prefetch(gA + kt + KC, 0, 1);
      __builtin_prefetch(gB + kt + KC, 0, 1);
    }
    wait_asynccnt0();   // this wave's DMA has landed in LDS
    __syncthreads();    // all waves' tiles visible

    // Keep A fragments resident; stream B fragments to limit VGPR pressure.
    v16h af0 = lds_frag(&sA[(wm * 32 +  0) * LDSS], LDSS);
    v16h af1 = lds_frag(&sA[(wm * 32 + 16) * LDSS], LDSS);
#pragma unroll
    for (int sn = 0; sn < 4; ++sn) {
      v16h bf = lds_frag(&sB[(wn * 64 + sn * 16) * LDSS], LDSS);
      acc[0][sn] = wmma_f32_f16(af0, bf, acc[0][sn]);
      acc[1][sn] = wmma_f32_f16(af1, bf, acc[1][sn]);
    }
    __syncthreads();    // consumers done before next chunk is staged
  }

  // Epilogue. C layout: VGPR i -> row i (lanes<16) / i+8 (lanes>=16);
  // col = lane & 15.
  const int rsel = (lane < 16) ? 0 : 8;
  const int ccol = lane & 15;
  const int cm = bm + wm * 32;
  const int cn = bn + wn * 64;

#pragma unroll
  for (int sm = 0; sm < 2; ++sm) {
#pragma unroll
    for (int sn = 0; sn < 4; ++sn) {
      const int col  = cn + sn * 16 + ccol;
      const float b  = bias ? bias[col] : 0.0f;
      const int row0 = cm + sm * 16 + rsel;
#pragma unroll
      for (int i = 0; i < 8; ++i) {
        float v = acc[sm][sn][i] + b;
        if (relu) v = fmaxf(v, 0.0f);
        const size_t off = (size_t)(row0 + i) * Nc + col;
        if (C32) C32[off] = v;
        if (C16) C16[off] = (_Float16)v;
      }
    }
  }
}

// ---------------------------------------------------------------------------
// Small support kernels
// ---------------------------------------------------------------------------
__global__ void k_f32_to_f16(const float* __restrict__ s,
                             _Float16* __restrict__ d, int n) {
  int i = blockIdx.x * blockDim.x + threadIdx.x;
  if (i < n) d[i] = (_Float16)s[i];
}

// Pad I_w (rows x kin) to (rows x kpad) with zeros, converting to f16.
__global__ void k_pad_w(const float* __restrict__ s, _Float16* __restrict__ d,
                        int rows, int kin, int kpad) {
  int i = blockIdx.x * blockDim.x + threadIdx.x;
  if (i >= rows * kpad) return;
  int r = i / kpad, c = i - r * kpad;
  d[i] = (c < kin) ? (_Float16)s[r * kin + c] : (_Float16)0.0f;
}

__global__ void k_zero32(float* __restrict__ p, int n) {
  int i = blockIdx.x * blockDim.x + threadIdx.x;
  if (i < n) p[i] = 0.0f;
}

// cat([x, mel, zero-pad]) as f16:  (NT, KIN)
__global__ void k_build_xcat(const float* __restrict__ x,
                             const float* __restrict__ mel,
                             _Float16* __restrict__ d) {
  int i = blockIdx.x * blockDim.x + threadIdx.x;
  if (i >= NT * KIN) return;
  int nt = i / KIN, c = i - nt * KIN;
  float v;
  if (c == 0)            v = x[nt];
  else if (c <= FEAT)    v = mel[(size_t)nt * FEAT + (c - 1)];
  else                   v = 0.0f;
  d[i] = (_Float16)v;
}

// GRU cell elementwise: gate order (r, z, n), PyTorch semantics.
//   G1 = x_t@Wih^T + bih, G2 = h@Whh^T + bhh, both (T, 3H).
//   Writes new h (f32 + f16) and residual-added layer output (f32 + f16).
__global__ void k_gru_cell(const float* __restrict__ G1,
                           const float* __restrict__ G2,
                           const float* __restrict__ res,
                           float* __restrict__ h32,
                           _Float16* __restrict__ h16,
                           float* __restrict__ xo32,
                           _Float16* __restrict__ xo16) {
  int idx = blockIdx.x * blockDim.x + threadIdx.x;
  if (idx >= T_BATCH * HID) return;
  int t = idx >> 9;          // / HID
  int j = idx & (HID - 1);   // % HID
  size_t g = (size_t)t * G3 + j;
  float g1r = G1[g], g1z = G1[g + HID], g1n = G1[g + 2 * HID];
  float g2r = G2[g], g2z = G2[g + HID], g2n = G2[g + 2 * HID];
  float r = 1.0f / (1.0f + expf(-(g1r + g2r)));
  float z = 1.0f / (1.0f + expf(-(g1z + g2z)));
  float n = tanhf(g1n + r * g2n);
  float hp = h32[idx];
  float hn = (1.0f - z) * n + z * hp;
  h32[idx] = hn;
  h16[idx] = (_Float16)hn;
  float xo = hn + res[idx];
  xo32[idx] = xo;
  xo16[idx] = (_Float16)xo;
}

// ---------------------------------------------------------------------------
// Orchestration
// ---------------------------------------------------------------------------
extern "C" void kernel_launch(void* const* d_in, const int* in_sizes, int n_in,
                              void* d_out, int out_size, void* d_ws, size_t ws_size,
                              hipStream_t stream) {
  (void)in_sizes; (void)n_in; (void)out_size; (void)ws_size;

  const float* x      = (const float*)d_in[0];
  const float* mel    = (const float*)d_in[1];
  const float* I_w    = (const float*)d_in[2];
  const float* I_b    = (const float*)d_in[3];
  const float* H1_wih = (const float*)d_in[4];
  const float* H1_whh = (const float*)d_in[5];
  const float* H1_bih = (const float*)d_in[6];
  const float* H1_bhh = (const float*)d_in[7];
  const float* H2_wih = (const float*)d_in[8];
  const float* H2_whh = (const float*)d_in[9];
  const float* H2_bih = (const float*)d_in[10];
  const float* H2_bhh = (const float*)d_in[11];
  const float* O1_w   = (const float*)d_in[12];
  const float* O1_b   = (const float*)d_in[13];
  const float* O2_w   = (const float*)d_in[14];
  const float* O2_b   = (const float*)d_in[15];
  float* out = (float*)d_out;   // (N, T, NCLASS) fp32

  // Bump allocator over the workspace (~830 MB; MI455X has 432 GiB).
  char* wp = (char*)d_ws;
  auto alloc = [&](size_t bytes) -> void* {
    void* r = (void*)wp;
    wp += (bytes + 255) & ~(size_t)255;
    return r;
  };
  _Float16* wI   = (_Float16*)alloc((size_t)HID * KIN * 2);
  _Float16* w1i  = (_Float16*)alloc((size_t)G3 * HID * 2);
  _Float16* w1h  = (_Float16*)alloc((size_t)G3 * HID * 2);
  _Float16* w2i  = (_Float16*)alloc((size_t)G3 * HID * 2);
  _Float16* w2h  = (_Float16*)alloc((size_t)G3 * HID * 2);
  _Float16* wo1  = (_Float16*)alloc((size_t)HID * HID * 2);
  _Float16* wo2  = (_Float16*)alloc((size_t)NCLASS * HID * 2);
  _Float16* xcat = (_Float16*)alloc((size_t)NT * KIN * 2);
  float*    xi32 = (float*)   alloc((size_t)NT * HID * 4);
  _Float16* xi16 = (_Float16*)alloc((size_t)NT * HID * 2);
  float*    h32  = (float*)   alloc((size_t)T_BATCH * HID * 4);
  _Float16* h16  = (_Float16*)alloc((size_t)T_BATCH * HID * 2);
  float*    G1   = (float*)   alloc((size_t)T_BATCH * G3 * 4);
  float*    G2   = (float*)   alloc((size_t)T_BATCH * G3 * 4);
  float*    xo32 = (float*)   alloc((size_t)NT * HID * 4);
  _Float16* xo16 = (_Float16*)alloc((size_t)NT * HID * 2);
  float*    yo32 = (float*)   alloc((size_t)NT * HID * 4);
  _Float16* yo16 = (_Float16*)alloc((size_t)NT * HID * 2);
  _Float16* y116 = (_Float16*)alloc((size_t)NT * HID * 2);

  const int TB = 256;
  auto blk = [](long n, int tb) { return (unsigned)((n + tb - 1) / tb); };

  // 1) Weight conversion (once per launch; deterministic).
  k_pad_w<<<blk((long)HID * KIN, TB), TB, 0, stream>>>(I_w, wI, HID, FEAT + 1, KIN);
  k_f32_to_f16<<<blk((long)G3 * HID, TB), TB, 0, stream>>>(H1_wih, w1i, G3 * HID);
  k_f32_to_f16<<<blk((long)G3 * HID, TB), TB, 0, stream>>>(H1_whh, w1h, G3 * HID);
  k_f32_to_f16<<<blk((long)G3 * HID, TB), TB, 0, stream>>>(H2_wih, w2i, G3 * HID);
  k_f32_to_f16<<<blk((long)G3 * HID, TB), TB, 0, stream>>>(H2_whh, w2h, G3 * HID);
  k_f32_to_f16<<<blk((long)HID * HID, TB), TB, 0, stream>>>(O1_w, wo1, HID * HID);
  k_f32_to_f16<<<blk((long)NCLASS * HID, TB), TB, 0, stream>>>(O2_w, wo2, NCLASS * HID);

  // 2) Input projection: xi = [x|mel] @ I_w^T + I_b   (NT x 512, K=160)
  k_build_xcat<<<blk((long)NT * KIN, TB), TB, 0, stream>>>(x, mel, xcat);
  {
    dim3 g(HID / BN, NT / BM);
    k_gemm_f16<<<g, 256, 0, stream>>>(xcat, wI, I_b, xi32, xi16,
                                      NT, HID, KIN, /*relu=*/0);
  }

  // 3) GRU layers: scan over N=8 steps, batch T=8192.
  dim3 gGate(G3 / BN, T_BATCH / BM);           // 8192 x 1536, K=512
  const unsigned cellBlocks = blk((long)T_BATCH * HID, TB);

  // ---- layer 1 ----
  k_zero32<<<blk((long)T_BATCH * HID, TB), TB, 0, stream>>>(h32, T_BATCH * HID);
  k_zero32<<<blk((long)T_BATCH * HID / 2, TB), TB, 0, stream>>>((float*)h16,
                                                               T_BATCH * HID / 2);
  for (int n = 0; n < N_STEPS; ++n) {
    const size_t off = (size_t)n * T_BATCH * HID;
    k_gemm_f16<<<gGate, 256, 0, stream>>>(xi16 + off, w1i, H1_bih, G1, nullptr,
                                          T_BATCH, G3, HID, 0);
    k_gemm_f16<<<gGate, 256, 0, stream>>>(h16, w1h, H1_bhh, G2, nullptr,
                                          T_BATCH, G3, HID, 0);
    k_gru_cell<<<cellBlocks, TB, 0, stream>>>(G1, G2, xi32 + off, h32, h16,
                                              xo32 + off, xo16 + off);
  }

  // ---- layer 2 ----
  k_zero32<<<blk((long)T_BATCH * HID, TB), TB, 0, stream>>>(h32, T_BATCH * HID);
  k_zero32<<<blk((long)T_BATCH * HID / 2, TB), TB, 0, stream>>>((float*)h16,
                                                               T_BATCH * HID / 2);
  for (int n = 0; n < N_STEPS; ++n) {
    const size_t off = (size_t)n * T_BATCH * HID;
    k_gemm_f16<<<gGate, 256, 0, stream>>>(xo16 + off, w2i, H2_bih, G1, nullptr,
                                          T_BATCH, G3, HID, 0);
    k_gemm_f16<<<gGate, 256, 0, stream>>>(h16, w2h, H2_bhh, G2, nullptr,
                                          T_BATCH, G3, HID, 0);
    k_gru_cell<<<cellBlocks, TB, 0, stream>>>(G1, G2, xo32 + off, h32, h16,
                                              yo32 + off, yo16 + off);
  }

  // 4) Output head: relu(yo @ O1_w^T + b) -> relu(. @ O2_w^T + b) -> out
  {
    dim3 g1(HID / BN, NT / BM);
    k_gemm_f16<<<g1, 256, 0, stream>>>(yo16, wo1, O1_b, nullptr, y116,
                                       NT, HID, HID, /*relu=*/1);
    dim3 g2(NCLASS / BN, NT / BM);
    k_gemm_f16<<<g2, 256, 0, stream>>>(y116, wo2, O2_b, out, nullptr,
                                       NT, NCLASS, HID, /*relu=*/1);
  }
}
